// MSAM_78211354460764
// MI455X (gfx1250) — compile-verified, matrix operand order checked by reading, not run on hardware
//
#include <hip/hip_runtime.h>
#include <hip/hip_bf16.h>

typedef __attribute__((ext_vector_type(16))) __bf16 v16bf;
typedef __attribute__((ext_vector_type(8)))  __bf16 v8bf;
typedef __attribute__((ext_vector_type(8)))  float  v8f;

#define B_   16
#define C_   64
#define CA_  16
#define H_   256
#define W_   256
#define HW_  (H_*W_)
#define NTAP 81      // 9x9
#define NCHUNK 41    // ceil(81 taps / 2) ; each K-chunk = 2 taps x 16 ch = 32

// ws layout (byte offsets, all >=32B aligned)
#define OFF_POOLED   0          // 256 f32
#define OFF_KW       2048       // 16*400 f32
#define OFF_PACKW    28672      // 41*32*16 bf16
#define OFF_PACKA    73728      // 8*32*16 bf16
#define OFF_CONVOUT  81920      // 16*16*256*256 bf16 (~33.6 MB)

static __device__ inline v8bf zero_v8bf() {
    v8bf r;
    for (int i = 0; i < 8; ++i) r[i] = (__bf16)0.0f;
    return r;
}

#define SHUF16(lo, hi) __builtin_shufflevector(lo, hi, \
    0, 1, 2, 3, 4, 5, 6, 7, 8, 9, 10, 11, 12, 13, 14, 15)

// ---------------- Kernel 1: channel mean pool of F_att -------------------
__global__ __launch_bounds__(256) void pool_kernel(const float* __restrict__ x,
                                                   float* __restrict__ pooled) {
    int bc = blockIdx.x;             // 0..255 = (b, c<16)
    int b = bc >> 4, c = bc & 15;
    const float* p = x + ((size_t)(b * C_ + c)) * HW_;
    float s = 0.0f;
    for (int i = threadIdx.x; i < HW_; i += 256) s += p[i];
    __shared__ float red[256];
    red[threadIdx.x] = s;
    __syncthreads();
    for (int st = 128; st > 0; st >>= 1) {
        if (threadIdx.x < st) red[threadIdx.x] += red[threadIdx.x + st];
        __syncthreads();
    }
    if (threadIdx.x == 0) pooled[bc] = red[0] * (1.0f / HW_);
}

// ---------------- Kernel 2: kernel-generating MLP ------------------------
__global__ __launch_bounds__(256) void mlp_kernel(const float* __restrict__ pooled,
                                                  const float* __restrict__ w1,
                                                  const float* __restrict__ b1,
                                                  const float* __restrict__ w2,
                                                  const float* __restrict__ b2,
                                                  float* __restrict__ kw) {
    __shared__ float hsh[B_ * 8];
    int tid = threadIdx.x;
    if (tid < B_ * 8) {
        int b = tid >> 3, j = tid & 7;
        float a = b1[j];
        for (int c = 0; c < CA_; ++c) a += pooled[b * CA_ + c] * w1[j * CA_ + c];
        hsh[tid] = 0.5f * a * (1.0f + erff(a * 0.70710678118f));   // exact GELU
    }
    __syncthreads();
    for (int e = tid; e < B_ * 400; e += 256) {
        int b = e / 400, k = e - b * 400;
        float a = b2[k];
        for (int j = 0; j < 8; ++j) a += hsh[b * 8 + j] * w2[k * 8 + j];
        kw[e] = a;
    }
}

// ------- Kernel 3: pack conv + fusion weights into WMMA A-frag layout ----
// A-matrix (16-bit, 16x32) per-lane layout: row M = lane%16;
// element i -> klocal = (i<8 ? i : i+8) + (lane<16 ? 0 : 8)
__global__ __launch_bounds__(256) void pack_kernel(const float* __restrict__ lkw,
                                                   const float* __restrict__ fusw,
                                                   __bf16* __restrict__ packW,
                                                   __bf16* __restrict__ packA) {
    int e = blockIdx.x * 256 + threadIdx.x;
    if (e < NCHUNK * 512) {                       // 9x9 conv weights
        int i = e & 15, l = (e >> 4) & 31, q = e >> 9;
        int o = l & 15;
        int kl = (i < 8 ? i : i + 8) + (l < 16 ? 0 : 8);
        int tap = 2 * q + (kl >= 16 ? 1 : 0);     // K order: k = tap*16 + c
        int c = kl & 15;
        float v = (tap < NTAP) ? lkw[(o * CA_ + c) * NTAP + tap] : 0.0f;
        packW[e] = (__bf16)v;
    } else {                                      // 64x64 fusion weights
        int e2 = e - NCHUNK * 512;
        if (e2 < 8 * 512) {
            int i = e2 & 15, l = (e2 >> 4) & 31, mq = e2 >> 9;
            int m = mq >> 1, kc = mq & 1;
            int o = m * 16 + (l & 15);
            int kl = (i < 8 ? i : i + 8) + (l < 16 ? 0 : 8);
            int c = kc * 32 + kl;
            packA[e2] = (__bf16)fusw[o * C_ + c];
        }
    }
}

// ------- Kernel 4: 9x9 dense conv (WMMA implicit GEMM) + dynamic 5x5 -----
// One block = one (b, 16x16 output tile). LDS tile: 24x24 halo x 16 ch,
// channel-innermost bf16 so each B fragment = two ds_load_b128 with
// compile-time-constant offsets. K-chunk loop is outer with two row
// accumulators so each packed-A fragment is loaded from global only once.
__global__ __launch_bounds__(256) void conv_kernel(const float* __restrict__ x,
                                                   const float* __restrict__ kw,
                                                   const __bf16* __restrict__ packW,
                                                   __bf16* __restrict__ convout) {
    __shared__ __bf16 lin[24 * 24 * 16];
    __shared__ float  kwl[CA_ * 25];
    const int tx = blockIdx.x, ty = blockIdx.y, b = blockIdx.z;
    const int tid = threadIdx.x;

    for (int p = tid; p < 24 * 24; p += 256) {
        int ly = p / 24, lx = p - ly * 24;
        int gy = ty * 16 + ly - 4, gx = tx * 16 + lx - 4;
        bool inb = (gy >= 0) & (gy < H_) & (gx >= 0) & (gx < W_);
        size_t base = ((size_t)(b * C_)) * HW_ + (size_t)gy * W_ + gx;
        for (int c = 0; c < CA_; ++c) {
            float v = inb ? x[base + (size_t)c * HW_] : 0.0f;
            lin[p * 16 + c] = (__bf16)v;
        }
    }
    for (int i = tid; i < CA_ * 25; i += 256) kwl[i] = kw[b * 400 + i];
    __syncthreads();

    const int wave = tid >> 5, lane = tid & 31;
    const int n  = lane & 15;            // output pixel column (N)
    const int cb = (lane < 16) ? 0 : 8;  // channel base of B elements / M offset
    const int y0 = wave * 2;             // first of 2 output rows per wave

    // B-fragment base pointers (center pixel) for the two rows
    const __bf16* bb0 = &lin[((y0 + 4) * 24 + (n + 4)) * 16 + cb];
    const __bf16* bb1 = bb0 + 24 * 16;

    v8f acc0 = {}, acc1 = {};
    const __bf16* ap = packW + lane * 16;
#pragma unroll
    for (int q = 0; q < NCHUNK; ++q) {
        v16bf a = *(const v16bf*)(ap + q * 512);
        const int t0 = 2 * q, t1 = 2 * q + 1;
        const int o0 = ((t0 / 9 - 4) * 24 + (t0 % 9 - 4)) * 16;   // constant
        v8bf lo0 = *(const v8bf*)(bb0 + o0);
        v8bf lo1 = *(const v8bf*)(bb1 + o0);
        v8bf hi0, hi1;
        if (t1 < NTAP) {
            const int o1 = ((t1 / 9 - 4) * 24 + (t1 % 9 - 4)) * 16; // constant
            hi0 = *(const v8bf*)(bb0 + o1);
            hi1 = *(const v8bf*)(bb1 + o1);
        } else {
            hi0 = zero_v8bf();
            hi1 = hi0;
        }
        v16bf b0 = SHUF16(lo0, hi0);
        v16bf b1 = SHUF16(lo1, hi1);
        acc0 = __builtin_amdgcn_wmma_f32_16x16x32_bf16(
            false, a, false, b0, (short)0, acc0, false, false);
        acc1 = __builtin_amdgcn_wmma_f32_16x16x32_bf16(
            false, a, false, b1, (short)0, acc1, false, false);
    }

    // dynamic depthwise 5x5 (per-sample kernel) + store bf16
#pragma unroll
    for (int rr = 0; rr < 2; ++rr) {
        const int yy = y0 + rr;
        const v8f acc = rr ? acc1 : acc0;
        const __bf16* dwb = &lin[((yy + 4) * 24 + (n + 4)) * 16];
        const int gy = ty * 16 + yy, gx = tx * 16 + n;
#pragma unroll
        for (int r = 0; r < 8; ++r) {
            const int ch = r + cb;
            float s = acc[r];
#pragma unroll
            for (int t = 0; t < 25; ++t) {
                const int off = ((t / 5 - 2) * 24 + (t % 5 - 2)) * 16; // constant
                s += kwl[ch * 25 + t] * (float)dwb[off + ch];
            }
            convout[((size_t)(b * CA_ + ch)) * HW_ + (size_t)gy * W_ + gx] = (__bf16)s;
        }
    }
}

// ------- Kernel 5: rolls + 64x64 1x1 fusion GEMM (WMMA) + bias -----------
__global__ __launch_bounds__(256) void fuse_kernel(const float* __restrict__ x,
                                                   const __bf16* __restrict__ convout,
                                                   const __bf16* __restrict__ packA,
                                                   const float* __restrict__ fusb,
                                                   float* __restrict__ out) {
    __shared__ __bf16 lch[256 * 64];     // [pixel][channel] bf16, 32 KB
    const int tx = blockIdx.x, ty = blockIdx.y, b = blockIdx.z;
    const int tid = threadIdx.x;
    const int ly = tid >> 4, lx = tid & 15;
    const int gy = ty * 16 + ly, gx = tx * 16 + lx;

    for (int c = 0; c < C_; ++c) {       // apply jnp.roll at gather time
        int ys = gy, xs = gx;
        switch (c >> 3) {
            case 0:  ys = (gy - 1 + H_) & (H_ - 1); break;
            case 1:  ys = (gy + 1) & (H_ - 1);      break;
            case 2:  xs = (gx - 1 + W_) & (W_ - 1); break;
            case 3:  xs = (gx + 1) & (W_ - 1);      break;
            case 4:  ys = (gy - 2 + H_) & (H_ - 1); break;
            case 5:  ys = (gy + 2) & (H_ - 1);      break;
            case 6:  xs = (gx - 2 + W_) & (W_ - 1); break;
            default: xs = (gx + 2) & (W_ - 1);      break;
        }
        __bf16 v;
        if (c < CA_) v = convout[((size_t)(b * CA_ + c)) * HW_ + (size_t)ys * W_ + xs];
        else         v = (__bf16)x[((size_t)(b * C_ + c)) * HW_ + (size_t)ys * W_ + xs];
        lch[tid * 64 + c] = v;
    }
    __syncthreads();

    const int wave = tid >> 5, lane = tid & 31;
    const int n  = lane & 15;
    const int cb = (lane < 16) ? 0 : 8;

    // Hoist all 8 packed-A fragments (4 M-tiles x 2 K-chunks) into registers.
    v16bf afr[8];
#pragma unroll
    for (int j = 0; j < 8; ++j)
        afr[j] = *(const v16bf*)(packA + (j * 32 + lane) * 16);

#pragma unroll
    for (int rr = 0; rr < 2; ++rr) {
        const int yy = wave * 2 + rr;
        const int p = yy * 16 + n;
        v16bf bm[2];
#pragma unroll
        for (int kc = 0; kc < 2; ++kc) {
            const __bf16* base = &lch[p * 64 + kc * 32 + cb];
            v8bf lo = *(const v8bf*)base;
            v8bf hi = *(const v8bf*)(base + 16);
            bm[kc] = SHUF16(lo, hi);
        }
        const int gyy = ty * 16 + yy, gxx = tx * 16 + n;
#pragma unroll
        for (int m = 0; m < 4; ++m) {    // 64 output channels = 4 M-tiles
            v8f acc = {};
            acc = __builtin_amdgcn_wmma_f32_16x16x32_bf16(
                false, afr[m * 2 + 0], false, bm[0], (short)0, acc, false, false);
            acc = __builtin_amdgcn_wmma_f32_16x16x32_bf16(
                false, afr[m * 2 + 1], false, bm[1], (short)0, acc, false, false);
#pragma unroll
            for (int r = 0; r < 8; ++r) {
                int o = m * 16 + r + cb;
                out[((size_t)(b * C_ + o)) * HW_ + (size_t)gyy * W_ + gxx] =
                    acc[r] + fusb[o];
            }
        }
    }
}

extern "C" void kernel_launch(void* const* d_in, const int* in_sizes, int n_in,
                              void* d_out, int out_size, void* d_ws, size_t ws_size,
                              hipStream_t stream) {
    const float* x     = (const float*)d_in[0];
    const float* kg_w1 = (const float*)d_in[1];
    const float* kg_b1 = (const float*)d_in[2];
    const float* kg_w2 = (const float*)d_in[3];
    const float* kg_b2 = (const float*)d_in[4];
    const float* lk_w  = (const float*)d_in[5];
    const float* fus_w = (const float*)d_in[6];
    const float* fus_b = (const float*)d_in[7];
    float* out = (float*)d_out;

    char* ws = (char*)d_ws;
    float*  pooled  = (float*)(ws + OFF_POOLED);
    float*  kwbuf   = (float*)(ws + OFF_KW);
    __bf16* packW   = (__bf16*)(ws + OFF_PACKW);
    __bf16* packA   = (__bf16*)(ws + OFF_PACKA);
    __bf16* convout = (__bf16*)(ws + OFF_CONVOUT);

    pool_kernel<<<256, 256, 0, stream>>>(x, pooled);
    mlp_kernel<<<1, 256, 0, stream>>>(pooled, kg_w1, kg_b1, kg_w2, kg_b2, kwbuf);
    pack_kernel<<<(NCHUNK * 512 + 8 * 512 + 255) / 256, 256, 0, stream>>>(
        lk_w, fus_w, packW, packA);

    dim3 grid(W_ / 16, H_ / 16, B_);     // 16x16x16 = 4096 blocks
    conv_kernel<<<grid, 256, 0, stream>>>(x, kwbuf, packW, convout);
    fuse_kernel<<<grid, 256, 0, stream>>>(x, convout, packA, fus_b, out);
}
